// DlrmSmall_38079180046653
// MI455X (gfx1250) — matrix operand order repaired and checked by reading, not compiled, not run on hardware
//
#include <hip/hip_runtime.h>
#include <math.h>

typedef unsigned short u16;
typedef u16    v8u  __attribute__((ext_vector_type(8)));
typedef u16    v16u __attribute__((ext_vector_type(16)));
typedef __bf16 v16bf __attribute__((ext_vector_type(16)));
typedef float  v8f  __attribute__((ext_vector_type(8)));

#define BATCH 16384
#define EMB_DIM 128
#define N_TABLES 26
#define VOCAB 100000

// ---------- helpers ----------
__device__ __forceinline__ u16 f2bf(float f) {
    unsigned u = __builtin_bit_cast(unsigned, f);
    unsigned r = u + 0x7FFFu + ((u >> 16) & 1u);   // round-to-nearest-even
    return (u16)(r >> 16);
}
__device__ __forceinline__ float bf2f(u16 b) {
    return __builtin_bit_cast(float, ((unsigned)b) << 16);
}
__device__ __forceinline__ unsigned lds_addr_of(const void* p) {
    return (unsigned)(unsigned long long)(__attribute__((address_space(3))) const char*)p;
}
__device__ __forceinline__ v16u load_a_frag(const u16* pa, int k0, int hi) {
    // A-layout (contiguous, matches B): lane<16 -> K 0..15; lane>=16 -> K 16..31
    return *(const v16u*)(pa + k0 + hi * 16);
}
__device__ __forceinline__ v8f wmma_bf16(v16u a, v16u b, v8f c) {
    return __builtin_amdgcn_wmma_f32_16x16x32_bf16(
        false, __builtin_bit_cast(v16bf, a),
        false, __builtin_bit_cast(v16bf, b),
        (short)0, c, false, false);
}

// ---------- fp32 -> bf16 pack with K padding ----------
__global__ void pack_bf16_kernel(const float* __restrict__ src, u16* __restrict__ dst,
                                 int rows, int sk, int dk) {
    long i = (long)blockIdx.x * blockDim.x + threadIdx.x;
    if (i >= (long)rows * dk) return;
    int r = (int)(i / dk);
    int c = (int)(i - (long)r * dk);
    dst[i] = (c < sk) ? f2bf(src[(long)r * sk + c]) : (u16)0;
}

// ---------- bf16 WMMA GEMM:  C = relu(A[M,K] * B[N,K]^T + bias) ----------
// Register-blocked: each wave owns a 32x64 tile (2 M-frags x 4 N-frags, 8 accums).
// Block = 256 threads (8 waves) -> 256x64 block tile. grid = (N/64, M/256).
__global__ __launch_bounds__(256) void gemm_bf16_relu(
    const u16* __restrict__ A, int lda,
    const u16* __restrict__ B, int ldb,
    const float* __restrict__ bias,
    u16* __restrict__ C, int ldc, int K) {
    const int wave = threadIdx.x >> 5;
    const int lane = threadIdx.x & 31;
    const int m16  = lane & 15;
    const int hi   = lane >> 4;
    const long row0 = ((long)blockIdx.y * 8 + wave) * 32;
    const int  col0 = blockIdx.x * 64;

    const u16* pa0 = A + (row0 + m16) * (long)lda;
    const u16* pa1 = A + (row0 + 16 + m16) * (long)lda;
    const u16* pb  = B + (long)(col0 + m16) * ldb;
    const long ldb16 = (long)ldb * 16;

    v8f acc[2][4];
    #pragma unroll
    for (int i = 0; i < 2; ++i)
        #pragma unroll
        for (int j = 0; j < 4; ++j) acc[i][j] = (v8f){};

    for (int k0 = 0; k0 < K; k0 += 32) {
        v16u a0 = load_a_frag(pa0, k0, hi);
        v16u a1 = load_a_frag(pa1, k0, hi);
        v16u b[4];
        #pragma unroll
        for (int j = 0; j < 4; ++j)   // B-layout (32x16, contiguous, same as A): lane<16 -> K 0..15; lane>=16 -> K 16..31
            b[j] = *(const v16u*)(pb + j * ldb16 + k0 + hi * 16);
        #pragma unroll
        for (int j = 0; j < 4; ++j) {
            acc[0][j] = wmma_bf16(a0, b[j], acc[0][j]);
            acc[1][j] = wmma_bf16(a1, b[j], acc[1][j]);
        }
    }

    // D layout: lane l, vgpr r -> (M = (l>>4)*8 + r, N = l&15)
    #pragma unroll
    for (int j = 0; j < 4; ++j) {
        const float bn = bias[col0 + j * 16 + m16];
        #pragma unroll
        for (int i = 0; i < 2; ++i) {
            u16* pc = C + (row0 + i * 16 + hi * 8) * (long)ldc + col0 + j * 16 + m16;
            #pragma unroll
            for (int r = 0; r < 8; ++r) {
                float v = acc[i][j][r] + bn;
                v = v > 0.0f ? v : 0.0f;
                pc[(long)r * ldc] = f2bf(v);
            }
        }
    }
}

// ---------- embedding gather (async-to-LDS) + pairwise interaction ----------
// One batch row per wave, 2 waves per block (64 threads).
// 26 embedding rows streamed into LDS via global_load_async_to_lds_b128 (ASYNCcnt),
// converted to bf16 into T (32x128), then Z = T*T^T via 2x2 WMMA tiles.
// R[b] = [ x_bf16 (128) | Zflat lower-strict (351) | 0 pad ] width 480.
#define IW 2
__global__ __launch_bounds__(IW * 32) void interact_kernel(
    const u16* __restrict__ xbot,           // [BATCH,128] bf16
    const float* __restrict__ emb,          // [26,100000,128] fp32
    const int* __restrict__ lS_i,           // [26,BATCH]
    u16* __restrict__ R) {                  // [BATCH,480] bf16
    __shared__ float E[IW][N_TABLES][EMB_DIM];   // fp32 staging for async gather
    __shared__ u16   T[IW][32][EMB_DIM];
    __shared__ float Z[IW][32][32];
    const int wave = threadIdx.x >> 5;
    const int lane = threadIdx.x & 31;
    const long b = (long)blockIdx.x * IW + wave;

    // issue all 26 gathers asynchronously: each lane moves 16B of the 512B row
    #pragma unroll 4
    for (int t = 0; t < N_TABLES; ++t) {
        int idx = lS_i[t * BATCH + b];
        const float* g = emb + ((long)t * VOCAB + idx) * EMB_DIM + lane * 4;
        unsigned la = lds_addr_of(&E[wave][t][lane * 4]);
        asm volatile("global_load_async_to_lds_b128 %0, %1, off"
                     :: "v"(la), "v"(g) : "memory");
    }

    {   // stage bottom-MLP output row (already bf16) while gathers are in flight
        const u16* src = xbot + b * EMB_DIM;
        #pragma unroll
        for (int c = 0; c < 4; ++c) T[wave][0][lane * 4 + c] = src[lane * 4 + c];
    }
    for (int j = lane; j < 5 * EMB_DIM; j += 32)    // zero pad rows 27..31
        T[wave][27 + (j >> 7)][j & 127] = 0;

    asm volatile("s_wait_asynccnt 0x0" ::: "memory");

    #pragma unroll 2
    for (int t = 0; t < N_TABLES; ++t) {            // fp32 -> bf16 into T
        #pragma unroll
        for (int c = 0; c < 4; ++c)
            T[wave][1 + t][lane * 4 + c] = f2bf(E[wave][t][lane * 4 + c]);
    }
    __syncthreads();

    const int m16 = lane & 15, hi = lane >> 4;
    #pragma unroll
    for (int mi = 0; mi < 2; ++mi) {
        #pragma unroll
        for (int ni = 0; ni < 2; ++ni) {
            v8f acc = {};
            const u16* pa = &T[wave][mi * 16 + m16][0];
            const u16* pb = &T[wave][ni * 16 + m16][0];
            #pragma unroll
            for (int k0 = 0; k0 < EMB_DIM; k0 += 32) {
                v16u av = load_a_frag(pa, k0, hi);
                v16u bv = *(const v16u*)(pb + k0 + hi * 16);
                acc = wmma_bf16(av, bv, acc);
            }
            #pragma unroll
            for (int r = 0; r < 8; ++r)
                Z[wave][mi * 16 + hi * 8 + r][ni * 16 + m16] = acc[r];
        }
    }
    __syncthreads();

    u16* out = R + b * 480;
    #pragma unroll
    for (int c = 0; c < 4; ++c) out[lane * 4 + c] = T[wave][0][lane * 4 + c];
    for (int p = lane; p < 351; p += 32) {          // strict lower triangle, (1,0),(2,0),(2,1),...
        int i = (int)((1.0f + sqrtf(1.0f + 8.0f * (float)p)) * 0.5f);
        while ((i * (i + 1)) / 2 <= p) ++i;
        while ((i * (i - 1)) / 2 > p) --i;
        int j = p - (i * (i - 1)) / 2;
        out[128 + p] = f2bf(Z[wave][i][j]);
    }
    if (lane == 0) out[479] = 0;
}

// ---------- final 256 -> 1 layer (wave-reduction dot), writes fp32 d_out ----------
__global__ __launch_bounds__(256) void final_layer_kernel(
    const u16* __restrict__ A, const float* __restrict__ W,
    const float* __restrict__ bias, float* __restrict__ out) {
    const int row  = blockIdx.x * 8 + (threadIdx.x >> 5);
    const int lane = threadIdx.x & 31;
    const u16* p = A + (long)row * 256;
    float s = 0.0f;
    #pragma unroll
    for (int k = lane; k < 256; k += 32) s += bf2f(p[k]) * W[k];
    for (int off = 16; off; off >>= 1) s += __shfl_xor(s, off, 32);
    if (lane == 0) {
        float v = s + bias[0];
        out[row] = v > 0.0f ? v : 0.0f;
    }
}

// ---------- host ----------
static inline void launch_pack(const float* src, u16* dst, int rows, int sk, int dk,
                               hipStream_t stream) {
    long total = (long)rows * dk;
    pack_bf16_kernel<<<(int)((total + 255) / 256), 256, 0, stream>>>(src, dst, rows, sk, dk);
}

extern "C" void kernel_launch(void* const* d_in, const int* in_sizes, int n_in,
                              void* d_out, int out_size, void* d_ws, size_t ws_size,
                              hipStream_t stream) {
    const float* dense_x = (const float*)d_in[0];
    const float* emb     = (const float*)d_in[1];
    const float* bW0 = (const float*)d_in[2];   // (512,13)
    const float* bW1 = (const float*)d_in[3];   // (256,512)
    const float* bW2 = (const float*)d_in[4];   // (128,256)
    const float* bb0 = (const float*)d_in[5];
    const float* bb1 = (const float*)d_in[6];
    const float* bb2 = (const float*)d_in[7];
    const float* tW0 = (const float*)d_in[8];   // (1024,479)
    const float* tW1 = (const float*)d_in[9];   // (1024,1024)
    const float* tW2 = (const float*)d_in[10];  // (512,1024)
    const float* tW3 = (const float*)d_in[11];  // (256,512)
    const float* tW4 = (const float*)d_in[12];  // (1,256)
    const float* tb0 = (const float*)d_in[13];
    const float* tb1 = (const float*)d_in[14];
    const float* tb2 = (const float*)d_in[15];
    const float* tb3 = (const float*)d_in[16];
    const float* tb4 = (const float*)d_in[17];
    const int*  lS_i = (const int*)d_in[18];
    // d_in[19] = lS_o, unused

    char* ws = (char*)d_ws;
    size_t off = 0;
    auto alloc = [&](size_t bytes) -> void* {
        void* p = ws + off;
        off = (off + bytes + 255) & ~(size_t)255;
        return p;
    };
    u16* x0p  = (u16*)alloc((size_t)BATCH * 32 * 2);     // dense_x padded 13->32
    u16* wb0  = (u16*)alloc((size_t)512 * 32 * 2);
    u16* wb1  = (u16*)alloc((size_t)256 * 512 * 2);
    u16* wb2  = (u16*)alloc((size_t)128 * 256 * 2);
    u16* wt0  = (u16*)alloc((size_t)1024 * 480 * 2);     // 479 -> 480
    u16* wt1  = (u16*)alloc((size_t)1024 * 1024 * 2);
    u16* wt2  = (u16*)alloc((size_t)512 * 1024 * 2);
    u16* wt3  = (u16*)alloc((size_t)256 * 512 * 2);
    u16* xbot = (u16*)alloc((size_t)BATCH * 128 * 2);    // bottom-MLP output, bf16
    u16* Rbuf = (u16*)alloc((size_t)BATCH * 480 * 2);    // interaction output, padded
    u16* bufA = (u16*)alloc((size_t)BATCH * 1024 * 2);   // ping
    u16* bufB = (u16*)alloc((size_t)BATCH * 1024 * 2);   // pong
    (void)ws_size; (void)in_sizes; (void)n_in; (void)out_size;

    // pack weights / dense input to bf16 (K padded to mult of 32)
    launch_pack(dense_x, x0p, BATCH, 13, 32, stream);
    launch_pack(bW0, wb0, 512, 13, 32, stream);
    launch_pack(bW1, wb1, 256, 512, 512, stream);
    launch_pack(bW2, wb2, 128, 256, 256, stream);
    launch_pack(tW0, wt0, 1024, 479, 480, stream);
    launch_pack(tW1, wt1, 1024, 1024, 1024, stream);
    launch_pack(tW2, wt2, 512, 1024, 1024, stream);
    launch_pack(tW3, wt3, 256, 512, 512, stream);

    const int MB = BATCH / 256;   // gemm grid.y (256 rows per block)

    // bottom MLP: 32 -> 512 -> 256 -> 128
    gemm_bf16_relu<<<dim3(512 / 64, MB), 256, 0, stream>>>(x0p, 32,  wb0, 32,  bb0, bufA, 512, 32);
    gemm_bf16_relu<<<dim3(256 / 64, MB), 256, 0, stream>>>(bufA, 512, wb1, 512, bb1, bufB, 256, 512);
    gemm_bf16_relu<<<dim3(128 / 64, MB), 256, 0, stream>>>(bufB, 256, wb2, 256, bb2, xbot, 128, 256);

    // embedding gather (async-to-LDS) + pairwise interaction -> R (BATCH x 480)
    interact_kernel<<<BATCH / IW, IW * 32, 0, stream>>>(xbot, emb, lS_i, Rbuf);

    // top MLP: 480 -> 1024 -> 1024 -> 512 -> 256 -> 1
    gemm_bf16_relu<<<dim3(1024 / 64, MB), 256, 0, stream>>>(Rbuf, 480,  wt0, 480,  tb0, bufA, 1024, 480);
    gemm_bf16_relu<<<dim3(1024 / 64, MB), 256, 0, stream>>>(bufA, 1024, wt1, 1024, tb1, bufB, 1024, 1024);
    gemm_bf16_relu<<<dim3(512 / 64,  MB), 256, 0, stream>>>(bufB, 1024, wt2, 1024, tb2, bufA, 512, 1024);
    gemm_bf16_relu<<<dim3(256 / 64,  MB), 256, 0, stream>>>(bufA, 512,  wt3, 512,  tb3, bufB, 256, 512);
    final_layer_kernel<<<BATCH / 8, 256, 0, stream>>>(bufB, tW4, tb4, (float*)d_out);
}
